// Plain_bert_57861799412028
// MI455X (gfx1250) — matrix operand order repaired and checked by reading, not compiled, not run on hardware
//
#include <hip/hip_runtime.h>
#include <hip/hip_bf16.h>
#include <math.h>

// ---------------------------------------------------------------------------
// Types for CDNA5 WMMA
// ---------------------------------------------------------------------------
typedef __attribute__((ext_vector_type(16))) __bf16 v16bf;
typedef __attribute__((ext_vector_type(8)))  __bf16 v8bf;
typedef __attribute__((ext_vector_type(4)))  __bf16 v4bf;
typedef __attribute__((ext_vector_type(2)))  __bf16 v2bf;
typedef __attribute__((ext_vector_type(8)))  float  v8f;
typedef __attribute__((ext_vector_type(4)))  float  v4f;

#define D_MODEL 768
#define N_HEAD 12
#define DHEAD 64
#define FF_DIM 3072
#define N_LAYER 12
#define LS 2560           // longformer sequence length
#define PAD_ID 1

__device__ __forceinline__ float gelu_exact(float x) {
    return 0.5f * x * (1.0f + erff(x * 0.70710678118654752f));
}

// ---------------------------------------------------------------------------
// One-time weight prep: fp32 W[K][N] -> bf16 Wt[N][K] (transposed), per layer.
// 32x32 LDS tile, coalesced loads and packed-b32 coalesced stores.
// ---------------------------------------------------------------------------
__global__ __launch_bounds__(256) void convert_w_kernel(
    const float* __restrict__ in, __bf16* __restrict__ out,
    int K, int N, long inStride, long outStride)
{
    const int l = blockIdx.z;
    const float* src = in  + (long)l * inStride;
    __bf16*      dst = out + (long)l * outStride;
    __shared__ float tile[32][33];
    const int k0 = blockIdx.y * 32, n0 = blockIdx.x * 32;
    const int tid = threadIdx.x;
    #pragma unroll
    for (int t = 0; t < 4; ++t) {
        int idx = tid + t * 256;          // 0..1023
        int r = idx >> 5, c = idx & 31;   // r: k, c: n (coalesced)
        tile[r][c] = src[(long)(k0 + r) * N + n0 + c];
    }
    __syncthreads();
    #pragma unroll
    for (int t = 0; t < 2; ++t) {
        int idx = tid + t * 256;          // 0..511
        int r  = idx >> 4;                // 0..31 : n
        int c2 = (idx & 15) * 2;          // 0..30 : k pair (coalesced)
        v2bf pk;
        pk[0] = (__bf16)tile[c2][r];
        pk[1] = (__bf16)tile[c2 + 1][r];
        *(v2bf*)&dst[(long)(n0 + r) * K + k0 + c2] = pk;
    }
}

// ---------------------------------------------------------------------------
// GEMM: C[M,N] = A[M,K] @ W[K,N] + bias, optional exact-GELU epilogue.
// A fp32 row stride lda; Wt = bf16 weights pre-transposed [N][K];
// C fp32 row-major (ldc=N).
// bf16 WMMA 16x16x32. Block = 256 threads (8 waves), tile 128x64, BK=32.
// Wave (wm,wn): 2 A frags x 2 B frags -> 4 WMMAs / K-step.
// Software pipeline, LDS double-buffered, one barrier per K-step.
// B tile staged with CDNA5 async global->LDS DMA (ASYNCcnt), A tile via the
// register pipeline (needs fp32->bf16 conversion).
// Requires N % 64 == 0 and K % 32 == 0 (true for 768/2304/3072); M guarded.
// ---------------------------------------------------------------------------
template<int EPI>
__global__ __launch_bounds__(256) void gemm_bf16_wmma(
    const float* __restrict__ A, long lda,
    const __bf16* __restrict__ Wt,
    const float* __restrict__ bias,
    float* __restrict__ C,
    int M, int N, int K)
{
    const int BM = 128, BN = 64, BK = 32;
    __shared__ __bf16 As[2][BM][BK + 8];   // pad: row stride 80B (16B aligned)
    __shared__ __bf16 Bt[2][BN][BK + 8];

    const int tid  = threadIdx.x;
    const int wave = tid >> 5;
    const int lane = tid & 31;
    const int wm   = wave >> 1;            // 0..3 : 32-row band
    const int wn   = wave & 1;             // 0..1 : 32-col band
    const int half = lane >> 4;
    const int l16  = lane & 15;

    const int m0 = blockIdx.y * BM;
    const int n0 = blockIdx.x * BN;

    v8f acc00 = {0.f,0.f,0.f,0.f,0.f,0.f,0.f,0.f};
    v8f acc01 = acc00, acc10 = acc00, acc11 = acc00;

    // ---- A staging (global fp32 -> regs -> bf16 LDS) ----
    auto loadA = [&](int kk, v4f ar[4]) {
        #pragma unroll
        for (int t = 0; t < 4; ++t) {
            int idx = tid + t * 256;           // 0..1023
            int r   = idx >> 3;                // 0..127
            int c4  = (idx & 7) * 4;           // 0..28
            int gm  = m0 + r;
            int gmc = (gm < M) ? gm : (M - 1);
            v4f v   = *(const v4f*)(A + (long)gmc * lda + kk + c4);
            float sc = (gm < M) ? 1.f : 0.f;
            ar[t] = v * sc;
        }
    };
    auto storeA = [&](int bi, const v4f ar[4]) {
        #pragma unroll
        for (int t = 0; t < 4; ++t) {
            int idx = tid + t * 256;
            int r   = idx >> 3;
            int c4  = (idx & 7) * 4;
            v4bf pk;
            pk[0] = (__bf16)ar[t][0]; pk[1] = (__bf16)ar[t][1];
            pk[2] = (__bf16)ar[t][2]; pk[3] = (__bf16)ar[t][3];
            *(v4bf*)&As[bi][r][c4] = pk;
        }
    };
    // ---- B staging: async DMA global -> LDS, 16B per thread ----
    const int bn_ = tid >> 2;                  // 0..63
    const int bko = (tid & 3) * 8;             // 0,8,16,24
    auto asyncB = [&](int kk, int bi) {
        const __bf16* gp = Wt + (long)(n0 + bn_) * K + kk + bko;
        unsigned ldsAddr = (unsigned)(unsigned long long)&Bt[bi][bn_][bko];
        asm volatile("global_load_async_to_lds_b128 %0, %1, off"
                     :: "v"(ldsAddr), "v"(gp) : "memory");
    };

    // ---- prologue: stage tile 0 ----
    {
        v4f ar[4];
        loadA(0, ar);
        asyncB(0, 0);
        storeA(0, ar);
    }

    int buf = 0;
    for (int k0 = 0; k0 < K; k0 += BK) {
        asm volatile("s_wait_asynccnt 0x0" ::: "memory");
        __syncthreads();
        const bool haveNext = (k0 + BK) < K;   // uniform
        v4f arN[4];
        if (haveNext) { loadA(k0 + BK, arN); asyncB(k0 + BK, buf ^ 1); }

        // ---- compute current tile: 4 WMMAs ----
        const int arow0 = wm * 32 + l16;
        const int arow1 = arow0 + 16;
        v8bf a0l = *(const v8bf*)&As[buf][arow0][8 * half];
        v8bf a0h = *(const v8bf*)&As[buf][arow0][16 + 8 * half];
        v8bf a1l = *(const v8bf*)&As[buf][arow1][8 * half];
        v8bf a1h = *(const v8bf*)&As[buf][arow1][16 + 8 * half];
        v16bf af0 = __builtin_shufflevector(a0l, a0h,
                        0,1,2,3,4,5,6,7,8,9,10,11,12,13,14,15);
        v16bf af1 = __builtin_shufflevector(a1l, a1h,
                        0,1,2,3,4,5,6,7,8,9,10,11,12,13,14,15);

        const int bc0 = wn * 32 + l16;
        const int bc1 = bc0 + 16;
        v8bf b0l = *(const v8bf*)&Bt[buf][bc0][8 * half];
        v8bf b0h = *(const v8bf*)&Bt[buf][bc0][16 + 8 * half];
        v8bf b1l = *(const v8bf*)&Bt[buf][bc1][8 * half];
        v8bf b1h = *(const v8bf*)&Bt[buf][bc1][16 + 8 * half];
        v16bf bf0 = __builtin_shufflevector(b0l, b0h,
                        0,1,2,3,4,5,6,7,8,9,10,11,12,13,14,15);
        v16bf bf1 = __builtin_shufflevector(b1l, b1h,
                        0,1,2,3,4,5,6,7,8,9,10,11,12,13,14,15);

        acc00 = __builtin_amdgcn_wmma_f32_16x16x32_bf16(
                    false, af0, false, bf0, (short)0, acc00, false, false);
        acc01 = __builtin_amdgcn_wmma_f32_16x16x32_bf16(
                    false, af0, false, bf1, (short)0, acc01, false, false);
        acc10 = __builtin_amdgcn_wmma_f32_16x16x32_bf16(
                    false, af1, false, bf0, (short)0, acc10, false, false);
        acc11 = __builtin_amdgcn_wmma_f32_16x16x32_bf16(
                    false, af1, false, bf1, (short)0, acc11, false, false);

        // ---- store next A tile into alternate buffer ----
        if (haveNext) { storeA(buf ^ 1, arN); }
        buf ^= 1;
    }

    // ---- epilogue ----
    const int mbase = m0 + wm * 32 + 8 * half;
    const int nbase = n0 + wn * 32 + l16;
    const float bias0 = bias[nbase];
    const float bias1 = bias[nbase + 16];
    const bool fullTile = (m0 + BM) <= M;     // uniform

    #pragma unroll
    for (int mi = 0; mi < 2; ++mi) {
        v8f a0 = (mi == 0) ? acc00 : acc10;
        v8f a1 = (mi == 0) ? acc01 : acc11;
        const int mrow = mbase + mi * 16;
        if (fullTile) {
            float* cp = C + (long)mrow * N + nbase;
            #pragma unroll
            for (int r = 0; r < 8; ++r) {
                float vA = a0[r] + bias0;
                float vB = a1[r] + bias1;
                if (EPI == 1) { vA = gelu_exact(vA); vB = gelu_exact(vB); }
                cp[0]  = vA;
                cp[16] = vB;
                cp += N;
            }
        } else {
            #pragma unroll
            for (int r = 0; r < 8; ++r) {
                int m = mrow + r;
                if (m < M) {
                    float vA = a0[r] + bias0;
                    float vB = a1[r] + bias1;
                    if (EPI == 1) { vA = gelu_exact(vA); vB = gelu_exact(vB); }
                    C[(long)m * N + nbase]      = vA;
                    C[(long)m * N + nbase + 16] = vB;
                }
            }
        }
    }
}

// ---------------------------------------------------------------------------
// Positions: pos = cumsum(mask)*mask + PAD, one wave32 per row (ballot scan)
// ---------------------------------------------------------------------------
__global__ __launch_bounds__(32) void pos_kernel(
    const int* __restrict__ ids, int* __restrict__ pos, int* __restrict__ mask, int S)
{
    const int row  = blockIdx.x;
    const int lane = threadIdx.x;
    int running = 0;
    for (int base = 0; base < S; base += 32) {
        long t = (long)row * S + base + lane;
        int id = ids[t];
        int mi = (id != PAD_ID) ? 1 : 0;
        unsigned bits = (unsigned)__ballot(mi);
        int incl = __popc(bits & (0xFFFFFFFFu >> (31 - lane)));
        pos[t]  = (running + incl) * mi + PAD_ID;
        mask[t] = mi;
        running += __popc(bits);
    }
}

// ---------------------------------------------------------------------------
// Embedding gather: X[t,:] = we[id] + pe[pos]
// ---------------------------------------------------------------------------
__global__ __launch_bounds__(256) void embed_kernel(
    const int* __restrict__ ids, const int* __restrict__ pos,
    const float* __restrict__ we, const float* __restrict__ pe,
    float* __restrict__ X)
{
    const long t = blockIdx.x;
    const int id = ids[t];
    const int p  = pos[t];
    #pragma unroll
    for (int c = 0; c < 3; ++c) {
        int i = c * 256 + threadIdx.x;
        X[t * D_MODEL + i] = we[(long)id * D_MODEL + i] + pe[(long)p * D_MODEL + i];
    }
}

// ---------------------------------------------------------------------------
// (optional residual add) + LayerNorm over D=768, one block per row
// ---------------------------------------------------------------------------
__global__ __launch_bounds__(256) void add_ln_kernel(
    const float* __restrict__ X, const float* __restrict__ Rr,
    const float* __restrict__ s, const float* __restrict__ bb,
    float* __restrict__ Out)
{
    const int row = blockIdx.x;
    __shared__ float buf[D_MODEL];
    __shared__ float red[256];
    const int tid = threadIdx.x;
    float sum = 0.f, sq = 0.f;
    #pragma unroll
    for (int t = 0; t < 3; ++t) {
        int i = t * 256 + tid;
        float v = X[(long)row * D_MODEL + i];
        if (Rr) v += Rr[(long)row * D_MODEL + i];
        buf[i] = v; sum += v; sq += v * v;
    }
    red[tid] = sum; __syncthreads();
    for (int st = 128; st > 0; st >>= 1) { if (tid < st) red[tid] += red[tid + st]; __syncthreads(); }
    float mean = red[0] * (1.f / D_MODEL); __syncthreads();
    red[tid] = sq; __syncthreads();
    for (int st = 128; st > 0; st >>= 1) { if (tid < st) red[tid] += red[tid + st]; __syncthreads(); }
    float var  = red[0] * (1.f / D_MODEL) - mean * mean;
    float rstd = rsqrtf(var + 1e-5f);
    __syncthreads();
    #pragma unroll
    for (int t = 0; t < 3; ++t) {
        int i = t * 256 + tid;
        Out[(long)row * D_MODEL + i] = (buf[i] - mean) * rstd * s[i] + bb[i];
    }
}

// ---------------------------------------------------------------------------
// Dense attention: S=64. One block per (seq, head); thread = query row.
// ---------------------------------------------------------------------------
__global__ __launch_bounds__(64) void dense_attn_kernel(
    const float* __restrict__ Y, const int* __restrict__ mask,
    float* __restrict__ O)
{
    const int r = blockIdx.x, h = blockIdx.y;
    __shared__ float Qs[64][65], Ks[64][65], Vs[64][65];
    const int tid = threadIdx.x;
    for (int idx = tid; idx < 64 * 64; idx += 64) {
        int j = idx >> 6, d = idx & 63;
        long row = (long)(r * 64 + j) * 2304;
        Qs[j][d] = Y[row + h * 64 + d];
        Ks[j][d] = Y[row + 768  + h * 64 + d];
        Vs[j][d] = Y[row + 1536 + h * 64 + d];
    }
    __syncthreads();
    const int qi = tid;
    float s[64];
    float m = -1e30f;
    for (int j = 0; j < 64; ++j) {
        float acc = 0.f;
        #pragma unroll 8
        for (int d = 0; d < 64; ++d) acc += Qs[qi][d] * Ks[j][d];
        acc = acc * 0.125f + (mask[r * 64 + j] ? 0.f : -1e9f);
        s[j] = acc; m = fmaxf(m, acc);
    }
    float l = 0.f;
    for (int j = 0; j < 64; ++j) { s[j] = __expf(s[j] - m); l += s[j]; }
    float inv = 1.f / l;
    for (int d = 0; d < 64; ++d) {
        float acc = 0.f;
        #pragma unroll 8
        for (int j = 0; j < 64; ++j) acc += s[j] * Vs[j][d];
        O[(long)(r * 64 + qi) * D_MODEL + h * 64 + d] = acc * inv;
    }
}

// ---------------------------------------------------------------------------
// Longformer windowed attention: block per (chunk n, head, batch row).
// ---------------------------------------------------------------------------
__global__ __launch_bounds__(64) void lf_window_kernel(
    const float* __restrict__ Y, const int* __restrict__ mask,
    float* __restrict__ O)
{
    const int n = blockIdx.x, h = blockIdx.y, b = blockIdx.z;
    __shared__ __bf16 Kw[192][64];
    __shared__ __bf16 Vw[192][64];
    __shared__ float k0s[64], v0s[64];
    const int tid = threadIdx.x;
    for (int idx = tid; idx < 192 * 64; idx += 64) {
        int j = idx >> 6, d = idx & 63;
        int kpos = (n - 1) * 64 + j;
        float kv = 0.f, vv = 0.f;
        if (kpos >= 0 && kpos < LS) {
            long row = ((long)b * LS + kpos) * 2304;
            kv = Y[row + 768  + h * 64 + d];
            vv = Y[row + 1536 + h * 64 + d];
        }
        Kw[j][d] = (__bf16)kv;
        Vw[j][d] = (__bf16)vv;
    }
    if (tid < 64) {
        long row0 = (long)b * LS * 2304;
        k0s[tid] = Y[row0 + 768  + h * 64 + tid];
        v0s[tid] = Y[row0 + 1536 + h * 64 + tid];
    }
    __syncthreads();

    const int qi = tid;
    long qrow = ((long)b * LS + n * 64 + qi) * 2304;
    float q[64];
    #pragma unroll 8
    for (int d = 0; d < 64; ++d) q[d] = Y[qrow + h * 64 + d];

    float m = -1e30f, l = 0.f, o[64];
    #pragma unroll 8
    for (int d = 0; d < 64; ++d) o[d] = 0.f;

    for (int j = 0; j < 192; ++j) {
        int rel  = j - 64 - qi;
        int kpos = (n - 1) * 64 + j;
        bool valid = (rel >= -64) && (rel <= 64) && (kpos > 0) && (kpos < LS)
                     && (mask[b * LS + kpos] != 0);
        if (!valid) continue;
        float acc = 0.f;
        #pragma unroll 8
        for (int d = 0; d < 64; ++d) acc += q[d] * (float)Kw[j][d];
        float s = acc * 0.125f;
        if (s <= m) {
            float p = __expf(s - m); l += p;
            #pragma unroll 8
            for (int d = 0; d < 64; ++d) o[d] += p * (float)Vw[j][d];
        } else {
            float rs = __expf(m - s); l = l * rs + 1.f;
            #pragma unroll 8
            for (int d = 0; d < 64; ++d) o[d] = o[d] * rs + (float)Vw[j][d];
            m = s;
        }
    }
    // global key 0 (unmasked)
    {
        float acc = 0.f;
        #pragma unroll 8
        for (int d = 0; d < 64; ++d) acc += q[d] * k0s[d];
        float s = acc * 0.125f;
        if (s <= m) {
            float p = __expf(s - m); l += p;
            #pragma unroll 8
            for (int d = 0; d < 64; ++d) o[d] += p * v0s[d];
        } else {
            float rs = __expf(m - s); l = l * rs + 1.f;
            #pragma unroll 8
            for (int d = 0; d < 64; ++d) o[d] = o[d] * rs + v0s[d];
            m = s;
        }
    }
    float inv = 1.f / l;
    long orow = ((long)b * LS + n * 64 + qi) * D_MODEL;
    #pragma unroll 8
    for (int d = 0; d < 64; ++d) O[orow + h * 64 + d] = o[d] * inv;
}

// ---------------------------------------------------------------------------
// Longformer global query (token 0) full attention over all S keys.
// ---------------------------------------------------------------------------
__global__ __launch_bounds__(256) void lf_global_kernel(
    const float* __restrict__ Y, const int* __restrict__ mask,
    float* __restrict__ O)
{
    const int b = blockIdx.x, h = blockIdx.y;
    __shared__ float sc[LS];
    __shared__ float q0[64];
    __shared__ float red[256];
    const int tid = threadIdx.x;
    if (tid < 64) q0[tid] = Y[(long)b * LS * 2304 + h * 64 + tid];
    __syncthreads();

    float lmax = -1e30f;
    for (int k = tid; k < LS; k += 256) {
        long row = ((long)b * LS + k) * 2304;
        float acc = 0.f;
        #pragma unroll 8
        for (int d = 0; d < 64; ++d) acc += q0[d] * Y[row + 768 + h * 64 + d];
        float s = acc * 0.125f + (mask[b * LS + k] ? 0.f : -1e9f);
        sc[k] = s; lmax = fmaxf(lmax, s);
    }
    red[tid] = lmax; __syncthreads();
    for (int st = 128; st > 0; st >>= 1) { if (tid < st) red[tid] = fmaxf(red[tid], red[tid + st]); __syncthreads(); }
    float m = red[0]; __syncthreads();

    float ls = 0.f;
    for (int k = tid; k < LS; k += 256) { float p = __expf(sc[k] - m); sc[k] = p; ls += p; }
    red[tid] = ls; __syncthreads();
    for (int st = 128; st > 0; st >>= 1) { if (tid < st) red[tid] += red[tid + st]; __syncthreads(); }
    float inv = 1.f / red[0];
    __syncthreads();

    if (tid < 64) {
        int d = tid; float acc = 0.f;
        for (int k = 0; k < LS; ++k)
            acc += sc[k] * Y[((long)b * LS + k) * 2304 + 1536 + h * 64 + d];
        O[(long)b * LS * D_MODEL + h * 64 + d] = acc * inv;
    }
}

// ---------------------------------------------------------------------------
// Final scoring + NLL loss
// ---------------------------------------------------------------------------
__global__ __launch_bounds__(128) void loss_kernel(
    const float* __restrict__ hf, const float* __restrict__ cf,
    const int* __restrict__ label, float* __restrict__ out)
{
    __shared__ float res[4];
    const int tid  = threadIdx.x;
    const int wave = tid >> 5;
    const int lane = tid & 31;
    const int b = wave >> 1, c = wave & 1;
    float acc = 0.f;
    for (int i = lane; i < D_MODEL; i += 32)
        acc += hf[b * D_MODEL + i] * cf[(b * 2 + c) * D_MODEL + i];
    #pragma unroll
    for (int off = 16; off > 0; off >>= 1) acc += __shfl_xor(acc, off, 32);
    if (lane == 0) res[wave] = acc;
    __syncthreads();
    if (tid == 0) {
        float loss = 0.f;
        for (int bb = 0; bb < 2; ++bb) {
            float r0 = res[bb * 2], r1 = res[bb * 2 + 1];
            float mm = fmaxf(r0, r1);
            float lse = mm + logf(expf(r0 - mm) + expf(r1 - mm));
            float sel = (label[bb] != 0) ? r1 : r0;
            loss += lse - sel;
        }
        out[0] = loss;
    }
}

// ---------------------------------------------------------------------------
// Host-side driver
// ---------------------------------------------------------------------------
struct EncParams {
    const float *we, *pe, *es, *eb, *Wqkv, *bqkv, *Wo, *bo,
                *l1s, *l1b, *Wi, *bi, *Wo2, *bo2, *l2s, *l2b;
};
struct EncWeightsT {      // pre-transposed bf16 weights, [N][K] per layer
    __bf16 *WqkvT, *WoT, *WiT, *Wo2T;
};

static void prep_weights(hipStream_t stream, const EncParams& p, const EncWeightsT& w)
{
    convert_w_kernel<<<dim3(2304/32, 768/32,  N_LAYER), 256, 0, stream>>>(
        p.Wqkv, w.WqkvT, 768,  2304, (long)768*2304,  (long)2304*768);
    convert_w_kernel<<<dim3(768/32,  768/32,  N_LAYER), 256, 0, stream>>>(
        p.Wo,   w.WoT,   768,  768,  (long)768*768,   (long)768*768);
    convert_w_kernel<<<dim3(3072/32, 768/32,  N_LAYER), 256, 0, stream>>>(
        p.Wi,   w.WiT,   768,  3072, (long)768*3072,  (long)3072*768);
    convert_w_kernel<<<dim3(768/32,  3072/32, N_LAYER), 256, 0, stream>>>(
        p.Wo2,  w.Wo2T,  3072, 768,  (long)3072*768,  (long)768*3072);
}

static void run_encoder(hipStream_t stream, const EncParams& p, const EncWeightsT& w,
                        const int* ids, int R, int S, bool longformer,
                        float* X, float* Y, float* AT, float* A2, float* HB,
                        int* pos, int* maskb)
{
    const int M = R * S;
    pos_kernel<<<R, 32, 0, stream>>>(ids, pos, maskb, S);
    embed_kernel<<<M, 256, 0, stream>>>(ids, pos, p.we, p.pe, X);
    add_ln_kernel<<<M, 256, 0, stream>>>(X, nullptr, p.es, p.eb, X);

    const dim3 gQKV(2304 / 64, (M + 127) / 128);
    const dim3 gD  (768  / 64, (M + 127) / 128);
    const dim3 gFF (3072 / 64, (M + 127) / 128);

    for (int l = 0; l < N_LAYER; ++l) {
        const __bf16* WqkvT = w.WqkvT + (long)l * 2304 * 768;
        const __bf16* WoT   = w.WoT   + (long)l * 768  * 768;
        const __bf16* WiT   = w.WiT   + (long)l * 3072 * 768;
        const __bf16* Wo2T  = w.Wo2T  + (long)l * 768  * 3072;
        const float* bqkv = p.bqkv + (long)l * 2304;
        const float* bo   = p.bo   + (long)l * 768;
        const float* bi   = p.bi   + (long)l * 3072;
        const float* bo2  = p.bo2  + (long)l * 768;

        gemm_bf16_wmma<0><<<gQKV, 256, 0, stream>>>(X, 768, WqkvT, bqkv, Y, M, 2304, 768);

        if (longformer) {
            lf_window_kernel<<<dim3(LS / 64, N_HEAD, R), 64, 0, stream>>>(Y, maskb, AT);
            lf_global_kernel<<<dim3(R, N_HEAD), 256, 0, stream>>>(Y, maskb, AT);
        } else {
            dense_attn_kernel<<<dim3(R, N_HEAD), 64, 0, stream>>>(Y, maskb, AT);
        }

        gemm_bf16_wmma<0><<<gD, 256, 0, stream>>>(AT, 768, WoT, bo, A2, M, 768, 768);
        add_ln_kernel<<<M, 256, 0, stream>>>(X, A2, p.l1s + l * 768, p.l1b + l * 768, X);
        gemm_bf16_wmma<1><<<gFF, 256, 0, stream>>>(X, 768, WiT, bi, HB, M, 3072, 768);
        gemm_bf16_wmma<0><<<gD, 256, 0, stream>>>(HB, 3072, Wo2T, bo2, A2, M, 768, 3072);
        add_ln_kernel<<<M, 256, 0, stream>>>(X, A2, p.l2s + l * 768, p.l2b + l * 768, X);
    }
}

extern "C" void kernel_launch(void* const* d_in, const int* in_sizes, int n_in,
                              void* d_out, int out_size, void* d_ws, size_t ws_size,
                              hipStream_t stream)
{
    (void)in_sizes; (void)n_in; (void)out_size; (void)ws_size;

    const int* his_id = (const int*)d_in[0];
    const int* can_id = (const int*)d_in[1];
    const int* label  = (const int*)d_in[2];

    auto loadEnc = [&](int base) {
        EncParams p;
        p.we   = (const float*)d_in[base + 0];
        p.pe   = (const float*)d_in[base + 1];
        p.es   = (const float*)d_in[base + 2];
        p.eb   = (const float*)d_in[base + 3];
        p.Wqkv = (const float*)d_in[base + 4];
        p.bqkv = (const float*)d_in[base + 5];
        p.Wo   = (const float*)d_in[base + 6];
        p.bo   = (const float*)d_in[base + 7];
        p.l1s  = (const float*)d_in[base + 8];
        p.l1b  = (const float*)d_in[base + 9];
        p.Wi   = (const float*)d_in[base + 10];
        p.bi   = (const float*)d_in[base + 11];
        p.Wo2  = (const float*)d_in[base + 12];
        p.bo2  = (const float*)d_in[base + 13];
        p.l2s  = (const float*)d_in[base + 14];
        p.l2b  = (const float*)d_in[base + 15];
        return p;
    };
    EncParams sp = loadEnc(3);
    EncParams dp = loadEnc(19);
    const float* h_w   = (const float*)d_in[35];
    const float* h_b   = (const float*)d_in[36];
    const float* h_lns = (const float*)d_in[37];
    const float* h_lnb = (const float*)d_in[38];

    // Workspace layout
    const int MAXM = 2 * LS;  // 5120
    char*  ws  = (char*)d_ws;
    size_t off = 0;
    auto alloc = [&](size_t bytes) -> void* {
        void* p = ws + off;
        off += (bytes + 255) & ~(size_t)255;
        return p;
    };
    float* X   = (float*)alloc((size_t)MAXM * 768  * 4);
    float* Y   = (float*)alloc((size_t)MAXM * 2304 * 4);
    float* AT  = (float*)alloc((size_t)MAXM * 768  * 4);
    float* A2  = (float*)alloc((size_t)MAXM * 768  * 4);
    float* HB  = (float*)alloc((size_t)MAXM * 3072 * 4);
    int*   pos = (int*)  alloc((size_t)MAXM * 4);
    int*   msk = (int*)  alloc((size_t)MAXM * 4);
    float* PHr = (float*)alloc((size_t)2 * 768 * 4);
    float* PH  = (float*)alloc((size_t)2 * 768 * 4);
    float* PCr = (float*)alloc((size_t)4 * 768 * 4);
    float* PC  = (float*)alloc((size_t)4 * 768 * 4);

    // bf16 transposed weights
    EncWeightsT sw, dw;
    sw.WqkvT = (__bf16*)alloc((size_t)N_LAYER * 2304 * 768 * 2);
    sw.WoT   = (__bf16*)alloc((size_t)N_LAYER * 768  * 768 * 2);
    sw.WiT   = (__bf16*)alloc((size_t)N_LAYER * 3072 * 768 * 2);
    sw.Wo2T  = (__bf16*)alloc((size_t)N_LAYER * 768  * 3072 * 2);
    dw.WqkvT = (__bf16*)alloc((size_t)N_LAYER * 2304 * 768 * 2);
    dw.WoT   = (__bf16*)alloc((size_t)N_LAYER * 768  * 768 * 2);
    dw.WiT   = (__bf16*)alloc((size_t)N_LAYER * 3072 * 768 * 2);
    dw.Wo2T  = (__bf16*)alloc((size_t)N_LAYER * 768  * 3072 * 2);
    __bf16* HWt = (__bf16*)alloc((size_t)768 * 768 * 2);

    // ---- One-time weight conversion (this launch) ----
    prep_weights(stream, sp, sw);
    prep_weights(stream, dp, dw);
    convert_w_kernel<<<dim3(768/32, 768/32, 1), 256, 0, stream>>>(
        h_w, HWt, 768, 768, 0, 0);

    // ---- History encoder (longformer), 2 rows x 2560 tokens ----
    run_encoder(stream, sp, sw, his_id, 2, LS, true, X, Y, AT, A2, HB, pos, msk);
    gemm_bf16_wmma<0><<<dim3(768 / 64, 1), 256, 0, stream>>>(
        X, (long)LS * 768, HWt, h_b, PHr, 2, 768, 768);
    add_ln_kernel<<<2, 256, 0, stream>>>(PHr, nullptr, h_lns, h_lnb, PH);

    // ---- Candidate encoder (dense), 4 rows x 64 tokens ----
    run_encoder(stream, dp, dw, can_id, 4, 64, false, X, Y, AT, A2, HB, pos, msk);
    gemm_bf16_wmma<0><<<dim3(768 / 64, 1), 256, 0, stream>>>(
        X, (long)64 * 768, HWt, h_b, PCr, 4, 768, 768);
    add_ln_kernel<<<4, 256, 0, stream>>>(PCr, nullptr, h_lns, h_lnb, PC);

    // ---- Scoring + NLL loss ----
    loss_kernel<<<1, 128, 0, stream>>>(PH, PC, label, (float*)d_out);
}
